// PygHomoLinkPredictionModel_49306224558369
// MI455X (gfx1250) — matrix operand order repaired and checked by reading, not compiled, not run on hardware
//
#include <hip/hip_runtime.h>
#include <math.h>

// ---------------- problem constants (match reference) ----------------
#define NN 100000      // nodes
#define EE 1600000     // edges
#define ELN 200000     // edge-label pairs
#define HD 128         // feature / hidden width
#define HL 256         // LSTM hidden
#define G4 1024        // 4*HL
#define NLAYER 4       // layer stack depth (1 GCN + 3 SG)
#define EPS 1e-5f

typedef __attribute__((ext_vector_type(16))) _Float16 v16h;
typedef __attribute__((ext_vector_type(8)))  float    v8f;

union FragU { v16h v; unsigned u[8]; uint4 q[2]; };

__device__ __forceinline__ unsigned packf16(float a, float b) {
  union { _Float16 h; unsigned short u; } ua, ub;
  ua.h = (_Float16)a; ub.h = (_Float16)b;
  return (unsigned)ua.u | ((unsigned)ub.u << 16);
}
__device__ __forceinline__ float sigf(float x) { return 1.f / (1.f + __expf(-x)); }

__device__ __forceinline__ v8f wmma_f16(const FragU& a, const FragU& b, v8f c) {
  return __builtin_amdgcn_wmma_f32_16x16x32_f16(false, a.v, false, b.v, (short)0, c,
                                                false, false);
}
__device__ __forceinline__ void ld_frag_lds(FragU& f, const unsigned* p) {
  const uint4* q = (const uint4*)p;
  f.q[0] = q[0]; f.q[1] = q[1];
}
__device__ __forceinline__ void ld_frag_glb(FragU& f, const unsigned* p) {
  const uint4* q = (const uint4*)p;
  f.q[0] = q[0]; f.q[1] = q[1];
}
// fragment-order layout: the 8 dwords a lane needs for one 16x16x32 tile are stored
// contiguously at ((tile)*32 + lane)*8.  One tile = 256 dwords = 1024 bytes.
//   A dword p maps to K-dword kd = kt*16 + (hi<<2) + (p&3) + ((p>>2)<<3)
//   B dword v maps to K-dword kd = kt*16 + (hi<<2) + (v&3) + ((v>>2)<<3), column ct*16 + (lane&15)

// ---------------- utility kernels ----------------
__global__ void zero_kernel(float* p, long long cnt) {
  long long t = (long long)blockIdx.x * blockDim.x + threadIdx.x;
  long long stride = (long long)gridDim.x * blockDim.x;
  for (; t < cnt; t += stride) p[t] = 0.f;
}

// pack weights into FRAGMENT-ORDER pair-packed f16 (see layout comment above).
// trans==0: W is [K][Nw] row-major ; trans==1: W is [Nw][K] row-major (use W^T)
__global__ void pack_weight_frag(const float* __restrict__ W, int K, int Nw, int trans,
                                 unsigned* __restrict__ out) {
  int nct = Nw >> 4;
  int total = (K >> 1) * Nw;
  for (int t = blockIdx.x * blockDim.x + threadIdx.x; t < total; t += gridDim.x * blockDim.x) {
    int v = t & 7;
    int lane = (t >> 3) & 31;
    int tile = t >> 8;
    int ct = tile % nct, kt = tile / nct;
    int n = lane & 15, hi = lane >> 4;
    int kd = kt * 16 + (hi << 2) + (v & 3) + ((v >> 2) << 3);
    int k0 = kd * 2;
    int col = ct * 16 + n;
    float a = trans ? W[(size_t)col * K + k0]     : W[(size_t)k0 * Nw + col];
    float b = trans ? W[(size_t)col * K + k0 + 1] : W[(size_t)(k0 + 1) * Nw + col];
    out[t] = packf16(a, b);
  }
}

// stage one fp32 element-pair into fragment-order LDS (A-matrix layout)
__device__ __forceinline__ int a_frag_lds_index(int r, int kk, int nkt) {
  int rt = r >> 4, m = r & 15;
  int kt = kk >> 4, rem = kk & 15;
  int hi = (rem >> 2) & 1;
  int p = (rem & 3) + ((rem >= 8) ? 4 : 0);
  return (((rt * nkt + kt) * 32) + (m + (hi << 4))) * 8 + p;
}

// ---------------- degree / normalization ----------------
__global__ void deg_init(float* deg, int n) {
  int t = blockIdx.x * blockDim.x + threadIdx.x;
  if (t < n) deg[t] = 1.f;  // self-loop
}
__global__ void deg_count(const int* __restrict__ dst, float* deg, int e) {
  for (int t = blockIdx.x * blockDim.x + threadIdx.x; t < e; t += gridDim.x * blockDim.x)
    atomicAdd(&deg[dst[t]], 1.f);
}
__global__ void deg_finalize(float* deg, int n) {
  int t = blockIdx.x * blockDim.x + threadIdx.x;
  if (t < n) deg[t] = rsqrtf(deg[t]);
}

// ---------------- GCN propagation: out = D^-1/2 (A+I) D^-1/2 in (+bias) ---------------
__global__ void prop_init(const float* __restrict__ in, const float* __restrict__ dinv,
                          const float* __restrict__ bias, float* __restrict__ out, int n) {
  long long t = (long long)blockIdx.x * blockDim.x + threadIdx.x;
  long long total = (long long)n * HD;
  if (t >= total) return;
  int node = (int)(t >> 7), f = (int)(t & 127);
  float d = dinv[node];
  out[t] = d * d * in[t] + (bias ? bias[f] : 0.f);
}
__global__ void prop_scatter(const float* __restrict__ in, const int* __restrict__ src,
                             const int* __restrict__ dst, const float* __restrict__ dinv,
                             float* __restrict__ out, int e) {
  long long total = (long long)e * HD;
  long long stride = (long long)gridDim.x * blockDim.x;
  for (long long t = (long long)blockIdx.x * blockDim.x + threadIdx.x; t < total; t += stride) {
    int edge = (int)(t >> 7), f = (int)(t & 127);
    int s = src[edge], d = dst[edge];
    float nrm = dinv[s] * dinv[d];
    atomicAdd(&out[(size_t)d * HD + f], nrm * in[(size_t)s * HD + f]);
  }
}

// ---------------- batchnorm (two pass; relu applied before stats per reference) -------
__global__ void bn_stats(const float* __restrict__ h, long long rows, float* stats, int do_relu) {
  long long total = rows * HD;
  long long t = (long long)blockIdx.x * blockDim.x + threadIdx.x;
  long long stride = (long long)gridDim.x * blockDim.x;  // multiple of 128
  int ch = (int)(t & 127);
  float s = 0.f, q = 0.f;
  for (; t < total; t += stride) {
    float v = h[t];
    if (do_relu) v = fmaxf(v, 0.f);
    s += v; q += v * v;
  }
  atomicAdd(&stats[ch], s);
  atomicAdd(&stats[128 + ch], q);
}
__global__ void bn_apply(float* h, long long rows, const float* __restrict__ stats,
                         const float* __restrict__ gamma, const float* __restrict__ beta,
                         int do_relu) {
  long long total = rows * HD;
  long long stride = (long long)gridDim.x * blockDim.x;
  for (long long t = (long long)blockIdx.x * blockDim.x + threadIdx.x; t < total; t += stride) {
    int ch = (int)(t & 127);
    float mu  = stats[ch] / (float)rows;
    float var = stats[128 + ch] / (float)rows - mu * mu;
    float v = h[t];
    if (do_relu) v = fmaxf(v, 0.f);
    h[t] = (v - mu) * rsqrtf(var + EPS) * gamma[ch] + beta[ch];
  }
}

// ---------------- WMMA GEMM: C[M,128] = A[M,K] @ Bfrag (+bias)(relu), K in {128,256} --
// Bpk fragment-order packed f16. Block: 256 thr = 8 waves; tile 32x64 (waves 2x4).
// If gRow!=null (K==256): A(e,k) = k<128 ? Asrc[gRow[e]][k] : Asrc[gCol[e]][k-128]
template <int K>
__global__ __launch_bounds__(256) void gemm_wmma(
    const float* __restrict__ A, int M, const unsigned* __restrict__ Bpk,
    const float* __restrict__ bias, float* __restrict__ C,
    int relu, const int* __restrict__ gRow, const int* __restrict__ gCol) {
  constexpr int KD = K >> 1;
  constexpr int NKT = K >> 5;
  constexpr int NCT = 8;  // Nw = 128
  __shared__ unsigned ldsA[32 * KD];  // frag-order
  const int row0 = blockIdx.x * 32;
  const int colBase = blockIdx.y * 64;
  const int tid = threadIdx.x;

  for (int idx = tid; idx < 32 * KD; idx += 256) {
    int r = idx / KD, kk = idx - r * KD;
    int grow = row0 + r;
    float a0 = 0.f, a1 = 0.f;
    if (grow < M) {
      if (gRow) {
        int half = (kk >= 64);
        int node = half ? gCol[grow] : gRow[grow];
        int f = kk * 2 - half * 128;
        const float* p = A + (size_t)node * 128 + f;
        a0 = p[0]; a1 = p[1];
      } else {
        const float* p = A + (size_t)grow * K + kk * 2;
        a0 = p[0]; a1 = p[1];
      }
    }
    ldsA[a_frag_lds_index(r, kk, NKT)] = packf16(a0, a1);
  }
  __syncthreads();

  const int wave = tid >> 5, lane = tid & 31;
  const int wr = wave >> 2, wc = wave & 3;
  const int m = lane & 15, hi = lane >> 4;
  const int ct = (colBase >> 4) + wc;

  const unsigned* abase = ldsA + (wr * NKT) * 256 + lane * 8;
  const unsigned* bbase = Bpk + ((size_t)ct * 32 + lane) * 8;

  v8f acc = {0.f, 0.f, 0.f, 0.f, 0.f, 0.f, 0.f, 0.f};
#pragma unroll
  for (int kt = 0; kt < NKT; ++kt) {
    FragU a, b;
    ld_frag_lds(a, abase + kt * 256);
    ld_frag_glb(b, bbase + kt * (NCT * 256));
    acc = wmma_f16(a, b, acc);
  }

  const int ccol = colBase + wc * 16 + m;
  const float bs = bias ? bias[ccol] : 0.f;
#pragma unroll
  for (int i = 0; i < 8; ++i) {
    int row = row0 + wr * 16 + i + hi * 8;
    if (row < M) {
      float v = acc[i] + bs;
      if (relu) v = fmaxf(v, 0.f);
      C[(size_t)row * HD + ccol] = v;
    }
  }
}

// ---------------- fused LSTM step + attention logit -----------------------------------
// gates = xt@Wih^T + h@Whh^T + b ; i,f,g,o split; c' = sig(f)c + sig(i)tanh(g);
// h' = sig(o)tanh(c'); logits[row] += h' . attw
// Block: 256 thr = 8 waves, 32 rows. Wave w covers hidden cols [w*32, w*32+32)
// as 2 subtiles of 16; per subtile it holds 2 row-groups x 4 gates accumulators and
// reuses each B fragment for both row-groups (halves weight traffic).
// Gate loop stays unrolled (constant 16KB-stride immediates); the k-tile loops are
// FORCED ROLLED (#pragma unroll 1) with pointer bumps so live registers stay at
// ~100 VGPRs (64 acc + 16 A-frag + 8 B-frag) and nothing spills to scratch.
__global__ __launch_bounds__(256) void lstm_step_kernel(
    const float* __restrict__ xt, float* __restrict__ h, float* __restrict__ c,
    const unsigned* __restrict__ WihPk,   // frag order K=128,N=1024
    const unsigned* __restrict__ WhhPk,   // frag order K=256,N=1024
    const float* __restrict__ b_ih, const float* __restrict__ b_hh,
    const float* __restrict__ attw,       // 256 floats
    float* __restrict__ logits) {
  __shared__ unsigned ldsX[32 * 64];   // 32 rows, K=128 (nkt=4), frag order
  __shared__ unsigned ldsH[32 * 128];  // 32 rows, K=256 (nkt=8), frag order
  const int tid = threadIdx.x;
  const int row0 = blockIdx.x * 32;

  for (int idx = tid; idx < 32 * 64; idx += 256) {
    int r = idx >> 6, kk = idx & 63;
    const float* q = xt + (size_t)(row0 + r) * HD + kk * 2;
    ldsX[a_frag_lds_index(r, kk, 4)] = packf16(q[0], q[1]);
  }
  for (int idx = tid; idx < 32 * 128; idx += 256) {
    int r = idx >> 7, kk = idx & 127;
    const float* q = h + (size_t)(row0 + r) * HL + kk * 2;
    ldsH[a_frag_lds_index(r, kk, 8)] = packf16(q[0], q[1]);
  }
  __syncthreads();

  const int wave = tid >> 5, lane = tid & 31;
  const int m = lane & 15, hi = lane >> 4;

  for (int s = 0; s < 2; ++s) {
    const int ctb = wave * 2 + s;   // col-tile of hidden index, 0..15
    const int j0 = ctb * 16;

    v8f acc[2][4];
#pragma unroll
    for (int rt = 0; rt < 2; ++rt)
#pragma unroll
      for (int g = 0; g < 4; ++g)
        acc[rt][g] = (v8f){0.f, 0.f, 0.f, 0.f, 0.f, 0.f, 0.f, 0.f};

    // x @ Wih^T : K=128 -> 4 k-tiles; gate stride 4096 dwords (16KB immediates)
    {
      const unsigned* xp = ldsX + lane * 8;
      const unsigned* wp = WihPk + ((size_t)ctb * 32 + lane) * 8;
#pragma unroll 1
      for (int kt = 0; kt < 4; ++kt) {
        FragU a0, a1;
        ld_frag_lds(a0, xp);
        ld_frag_lds(a1, xp + 4 * 256);
#pragma unroll
        for (int g = 0; g < 4; ++g) {
          FragU b;
          ld_frag_glb(b, wp + g * 4096);
          acc[0][g] = wmma_f16(a0, b, acc[0][g]);
          acc[1][g] = wmma_f16(a1, b, acc[1][g]);
        }
        xp += 256;
        wp += 16384;
      }
    }
    // h @ Whh^T : K=256 -> 8 k-tiles
    {
      const unsigned* hp = ldsH + lane * 8;
      const unsigned* wp = WhhPk + ((size_t)ctb * 32 + lane) * 8;
#pragma unroll 1
      for (int kt = 0; kt < 8; ++kt) {
        FragU a0, a1;
        ld_frag_lds(a0, hp);
        ld_frag_lds(a1, hp + 8 * 256);
#pragma unroll
        for (int g = 0; g < 4; ++g) {
          FragU b;
          ld_frag_glb(b, wp + g * 4096);
          acc[0][g] = wmma_f16(a0, b, acc[0][g]);
          acc[1][g] = wmma_f16(a1, b, acc[1][g]);
        }
        hp += 256;
        wp += 16384;
      }
    }

    const int col = j0 + m;  // hidden index in [0,256)
    const float bi = b_ih[col]          + b_hh[col];
    const float bf = b_ih[col + HL]     + b_hh[col + HL];
    const float bg = b_ih[col + 2 * HL] + b_hh[col + 2 * HL];
    const float bo = b_ih[col + 3 * HL] + b_hh[col + 3 * HL];
    const float aw = attw[col];
#pragma unroll
    for (int rt = 0; rt < 2; ++rt) {
#pragma unroll
      for (int i = 0; i < 8; ++i) {
        const int row = row0 + rt * 16 + i + hi * 8;
        const float vi = acc[rt][0][i] + bi;
        const float vf = acc[rt][1][i] + bf;
        const float vg = acc[rt][2][i] + bg;
        const float vo = acc[rt][3][i] + bo;
        const size_t idx = (size_t)row * HL + col;
        const float cp = c[idx];
        const float cn = sigf(vf) * cp + sigf(vi) * tanhf(vg);
        const float hn = sigf(vo) * tanhf(cn);
        c[idx] = cn;
        h[idx] = hn;
        atomicAdd(&logits[row], hn * aw);
      }
    }
  }
}

// ---------------- JK attention softmax + combine --------------------------------------
__global__ void jk_kernel(const float* __restrict__ xs, const float* __restrict__ logits,
                          float* __restrict__ jk, int n) {
  long long t = (long long)blockIdx.x * blockDim.x + threadIdx.x;
  long long total = (long long)n * HD;
  if (t >= total) return;
  int node = (int)(t >> 7);
  float l0 = logits[node], l1 = logits[n + node];
  float l2 = logits[2 * (size_t)n + node], l3 = logits[3 * (size_t)n + node];
  float mx = fmaxf(fmaxf(l0, l1), fmaxf(l2, l3));
  float e0 = __expf(l0 - mx), e1 = __expf(l1 - mx), e2 = __expf(l2 - mx), e3 = __expf(l3 - mx);
  float inv = 1.f / (e0 + e1 + e2 + e3);
  long long ls = (long long)n * HD;
  jk[t] = (e0 * xs[t] + e1 * xs[ls + t] + e2 * xs[2 * ls + t] + e3 * xs[3 * ls + t]) * inv;
}

// ---------------- predictor tail: BN-apply + dot(pred2) + sigmoid, wave per edge ------
__global__ void pred_final(const float* __restrict__ z1, long long rows,
                           const float* __restrict__ stats, const float* __restrict__ gamma,
                           const float* __restrict__ beta, const float* __restrict__ w2,
                           const float* __restrict__ b2, float* __restrict__ out) {
  const int wave = threadIdx.x >> 5, lane = threadIdx.x & 31;
  const long long e = (long long)blockIdx.x * 8 + wave;
  if (e >= rows) return;
  float acc = 0.f;
#pragma unroll
  for (int q = 0; q < 4; ++q) {
    const int j = lane + q * 32;
    const float mu  = stats[j] / (float)rows;
    const float var = stats[128 + j] / (float)rows - mu * mu;
    const float bn  = (z1[e * HD + j] - mu) * rsqrtf(var + EPS) * gamma[j] + beta[j];
    acc += bn * w2[j];
  }
#pragma unroll
  for (int off = 16; off > 0; off >>= 1) acc += __shfl_xor(acc, off, 32);
  if (lane == 0) out[e] = 1.f / (1.f + __expf(-(acc + b2[0])));
}

// =====================================================================================
extern "C" void kernel_launch(void* const* d_in, const int* in_sizes, int n_in,
                              void* d_out, int out_size, void* d_ws, size_t ws_size,
                              hipStream_t stream) {
  (void)in_sizes; (void)n_in; (void)out_size; (void)ws_size;
  const int* edge_index = (const int*)d_in[0];
  const float* x        = (const float*)d_in[1];
  const int* eli        = (const int*)d_in[2];
  const float* gcn_w = (const float*)d_in[3];
  const float* gcn_b = (const float*)d_in[4];
  const float* sg_w  = (const float*)d_in[5];
  const float* sg_b  = (const float*)d_in[6];
  const float* bn_g  = (const float*)d_in[7];
  const float* bn_b  = (const float*)d_in[8];
  const float* w_ih_f = (const float*)d_in[9];
  const float* w_hh_f = (const float*)d_in[10];
  const float* b_ih_f = (const float*)d_in[11];
  const float* b_hh_f = (const float*)d_in[12];
  const float* w_ih_b = (const float*)d_in[13];
  const float* w_hh_b = (const float*)d_in[14];
  const float* b_ih_b = (const float*)d_in[15];
  const float* b_hh_b = (const float*)d_in[16];
  const float* att_w  = (const float*)d_in[17];
  // d_in[18] = att_b: constant across layers -> cancels in softmax over layers
  const float* pred1_w = (const float*)d_in[19];
  const float* pred1_b = (const float*)d_in[20];
  const float* pbn_g   = (const float*)d_in[21];
  const float* pbn_b   = (const float*)d_in[22];
  const float* pred2_w = (const float*)d_in[23];
  const float* pred2_b = (const float*)d_in[24];
  float* out = (float*)d_out;

  const int* src = edge_index;
  const int* dst = edge_index + EE;
  const int* erow = eli;
  const int* ecol = eli + ELN;

  // ---- workspace carve (f32 units, 256B aligned chunks) ----
  float* ws = (float*)d_ws;
  size_t off = 0;
  auto alloc = [&](size_t nf) { float* p = ws + off; off += (nf + 63) & ~(size_t)63; return p; };
  float* xs     = alloc((size_t)NLAYER * NN * HD);  // layer stack [4,N,128]
  float* tmp    = alloc((size_t)NN * HD);
  float* dinv   = alloc(NN);
  float* hbuf   = alloc((size_t)NN * HL);
  float* cbuf   = alloc((size_t)NN * HL);
  float* logits = alloc((size_t)NLAYER * NN);
  float* jk     = alloc((size_t)NN * HD);
  float* z1     = alloc((size_t)ELN * HD);
  float* stats  = alloc(256);
  unsigned* gcn_pk   = (unsigned*)alloc(64 * 128);
  unsigned* sg_pk    = (unsigned*)alloc(3 * 64 * 128);
  unsigned* wihf_pk  = (unsigned*)alloc(64 * 1024);
  unsigned* whhf_pk  = (unsigned*)alloc(128 * 1024);
  unsigned* wihb_pk  = (unsigned*)alloc(64 * 1024);
  unsigned* whhb_pk  = (unsigned*)alloc(128 * 1024);
  unsigned* pred1_pk = (unsigned*)alloc(128 * 128);

  // ---- pack all GEMM weights to fragment-order f16 ----
  pack_weight_frag<<<64, 256, 0, stream>>>(gcn_w, 128, 128, 0, gcn_pk);
  for (int i = 0; i < 3; ++i)
    pack_weight_frag<<<64, 256, 0, stream>>>(sg_w + (size_t)i * 128 * 128, 128, 128, 0,
                                             sg_pk + (size_t)i * 64 * 128);
  pack_weight_frag<<<256, 256, 0, stream>>>(w_ih_f, 128, 1024, 1, wihf_pk);
  pack_weight_frag<<<512, 256, 0, stream>>>(w_hh_f, 256, 1024, 1, whhf_pk);
  pack_weight_frag<<<256, 256, 0, stream>>>(w_ih_b, 128, 1024, 1, wihb_pk);
  pack_weight_frag<<<512, 256, 0, stream>>>(w_hh_b, 256, 1024, 1, whhb_pk);
  pack_weight_frag<<<128, 256, 0, stream>>>(pred1_w, 256, 128, 0, pred1_pk);

  // ---- degree -> D^-1/2 ----
  deg_init<<<(NN + 255) / 256, 256, 0, stream>>>(dinv, NN);
  deg_count<<<4096, 256, 0, stream>>>(dst, dinv, EE);
  deg_finalize<<<(NN + 255) / 256, 256, 0, stream>>>(dinv, NN);

  const dim3 gemmGrid(NN / 32, HD / 64);  // 3125 x 2
  const long long NF = (long long)NN * HD;
  const int nfBlocks = (int)((NF + 255) / 256);

  // ---- GCNConv: tmp = x @ gcn_w ; xs0 = prop(tmp) + gcn_b ; BN(relu(.)) ----
  gemm_wmma<128><<<gemmGrid, 256, 0, stream>>>(x, NN, gcn_pk, nullptr, tmp, 0,
                                               nullptr, nullptr);
  prop_init<<<nfBlocks, 256, 0, stream>>>(tmp, dinv, gcn_b, xs, NN);
  prop_scatter<<<8192, 256, 0, stream>>>(tmp, src, dst, dinv, xs, EE);
  zero_kernel<<<1, 256, 0, stream>>>(stats, 256);
  bn_stats<<<1024, 256, 0, stream>>>(xs, NN, stats, 1);
  bn_apply<<<4096, 256, 0, stream>>>(xs, NN, stats, bn_g, bn_b, 1);

  // ---- SGConv stack: xs[i+1] = prop(xs[i]) @ sg_w[i] + sg_b[i] ; BN(relu) except last
  for (int i = 0; i < 3; ++i) {
    float* hin = xs + (size_t)i * NN * HD;
    float* hout = xs + (size_t)(i + 1) * NN * HD;
    prop_init<<<nfBlocks, 256, 0, stream>>>(hin, dinv, nullptr, tmp, NN);
    prop_scatter<<<8192, 256, 0, stream>>>(hin, src, dst, dinv, tmp, EE);
    gemm_wmma<128><<<gemmGrid, 256, 0, stream>>>(tmp, NN, sg_pk + (size_t)i * 64 * 128,
                                                 sg_b + (size_t)i * HD, hout, 0, nullptr,
                                                 nullptr);
    if (i < 2) {
      zero_kernel<<<1, 256, 0, stream>>>(stats, 256);
      bn_stats<<<1024, 256, 0, stream>>>(hout, NN, stats, 1);
      bn_apply<<<4096, 256, 0, stream>>>(hout, NN, stats, bn_g, bn_b, 1);
    }
  }

  // ---- bidirectional JK-LSTM with fused attention logits ----
  zero_kernel<<<4096, 256, 0, stream>>>(logits, (long long)NLAYER * NN);
  // forward
  zero_kernel<<<8192, 256, 0, stream>>>(hbuf, (long long)NN * HL);
  zero_kernel<<<8192, 256, 0, stream>>>(cbuf, (long long)NN * HL);
  for (int t = 0; t < NLAYER; ++t)
    lstm_step_kernel<<<NN / 32, 256, 0, stream>>>(xs + (size_t)t * NN * HD, hbuf, cbuf,
                                                  wihf_pk, whhf_pk, b_ih_f, b_hh_f, att_w,
                                                  logits + (size_t)t * NN);
  // backward (time-reversed input, att_w[256:512] slice)
  zero_kernel<<<8192, 256, 0, stream>>>(hbuf, (long long)NN * HL);
  zero_kernel<<<8192, 256, 0, stream>>>(cbuf, (long long)NN * HL);
  for (int t = 0; t < NLAYER; ++t) {
    int layer = NLAYER - 1 - t;
    lstm_step_kernel<<<NN / 32, 256, 0, stream>>>(xs + (size_t)layer * NN * HD, hbuf, cbuf,
                                                  wihb_pk, whhb_pk, b_ih_b, b_hh_b,
                                                  att_w + HL, logits + (size_t)layer * NN);
  }

  // ---- softmax over layers + JK combine ----
  jk_kernel<<<nfBlocks, 256, 0, stream>>>(xs, logits, jk, NN);

  // ---- edge predictor: gather-fused GEMM -> relu -> BN -> dot -> sigmoid ----
  const dim3 predGrid(ELN / 32, HD / 64);  // 6250 x 2
  gemm_wmma<256><<<predGrid, 256, 0, stream>>>(jk, ELN, pred1_pk, pred1_b, z1, 1,
                                               erow, ecol);
  zero_kernel<<<1, 256, 0, stream>>>(stats, 256);
  bn_stats<<<1024, 256, 0, stream>>>(z1, ELN, stats, 0);
  pred_final<<<ELN / 8, 256, 0, stream>>>(z1, ELN, stats, pbn_g, pbn_b, pred2_w, pred2_b,
                                          out);
}
